// SelfAttention_26766236189055
// MI455X (gfx1250) — compile-verified, matrix operand order checked by reading, not run on hardware
//
#include <hip/hip_runtime.h>
#include <stdint.h>

// =====================================================================
// Self-attention (softmax over QUERY axis) + BatchNorm + residual
// MI455X / gfx1250: wave32, V_WMMA_F32_16X16X32_BF16 for all GEMMs,
// f32 accumulation. 128x128 block tiles, 8 waves x (64x32) per wave
// (4x2 frags, 64 acc VGPRs -> no scratch spills), double-buffered
// GLOBAL_LOAD_ASYNC_TO_LDS_B128 staging (ASYNCcnt), ds_load_b128
// fragment feeds.
// =====================================================================

#define DEV_INLINE __device__ __forceinline__

static constexpr int B_  = 8;
static constexpr int C_  = 1024;
static constexpr int T_  = 2048;
static constexpr int CI_ = 512;
static constexpr float BN_EPS_ = 1e-5f;

typedef __attribute__((ext_vector_type(16))) __bf16 v16bf;
typedef __attribute__((ext_vector_type(8)))  float  v8f;
typedef int v4i __attribute__((vector_size(16)));   // matches builtin param

// ---------------- async global->LDS plumbing ------------------------
#if defined(__AMDGCN__) && __has_builtin(__builtin_amdgcn_global_load_async_to_lds_b128)
#define HAS_ASYNC 1
#else
#define HAS_ASYNC 0
#endif

#if HAS_ASYNC
#if __has_builtin(__builtin_amdgcn_s_wait_asynccnt)
#define WAIT_ASYNC(n) __builtin_amdgcn_s_wait_asynccnt(n)
#else
#define WAIT_ASYNC(n) asm volatile("s_wait_asynccnt %0" ::"i"(n) : "memory")
#endif
#else
#define WAIT_ASYNC(n) do { } while (0)
#endif

// ---------------- workspace layout (bytes, all 256-aligned) ----------
static constexpr size_t OFF_XT   = 0;                                   // bf16 [B][T][C]
static constexpr size_t OFF_WQKV = OFF_XT   + (size_t)B_*T_*C_*2;       // bf16 [3CI][C]
static constexpr size_t OFF_WO   = OFF_WQKV + (size_t)3*CI_*C_*2;       // bf16 [C][CI]
static constexpr size_t OFF_Q    = OFF_WO   + (size_t)C_*CI_*2;         // bf16 [B][T][CI]
static constexpr size_t OFF_K    = OFF_Q    + (size_t)B_*T_*CI_*2;      // bf16 [B][T][CI]
static constexpr size_t OFF_VT   = OFF_K    + (size_t)B_*T_*CI_*2;      // bf16 [B][CI][T]
static constexpr size_t OFF_S    = OFF_VT   + (size_t)B_*CI_*T_*2;      // f32  [B][T][T]
static constexpr size_t OFF_CMAX = OFF_S    + (size_t)B_*T_*T_*4;       // f32  [B][T]
static constexpr size_t OFF_CINV = OFF_CMAX + (size_t)B_*T_*4;          // f32  [B][T]
static constexpr size_t OFF_Y    = OFF_CINV + (size_t)B_*T_*4;          // bf16 [B][T][CI]
static constexpr size_t OFF_WY   = OFF_Y    + (size_t)B_*T_*CI_*2;      // f32  [B][C][T]
static constexpr size_t OFF_MEAN = OFF_WY   + (size_t)B_*C_*T_*4;       // f32  [C]
static constexpr size_t OFF_RSTD = OFF_MEAN + (size_t)C_*4;             // f32  [C]
// total ~306.3 MB

// ---------------- helpers -------------------------------------------
DEV_INLINE unsigned short f32_to_bf16(float f) {
  union { float f; uint32_t u; } c; c.f = f;
  uint32_t u = c.u;
  u += 0x7FFFu + ((u >> 16) & 1u);            // round-to-nearest-even
  return (unsigned short)(u >> 16);
}

DEV_INLINE v8f wmma_bf16(v16bf a, v16bf b, v8f c) {
  return __builtin_amdgcn_wmma_f32_16x16x32_bf16(
      /*neg_a=*/false, a, /*neg_b=*/false, b,
      /*c_mod=*/(short)0, c, /*reuse_a=*/false, /*reuse_b=*/false);
}

// Load one 16x32 bf16 fragment from a row-major K-contiguous matrix in
// GLOBAL memory (used by gemm_pv's V operand).
DEV_INLINE v16bf load_frag_b(const unsigned short* base, int ld, int row0, int k0) {
  const int lane = threadIdx.x & 31;
  const int half = lane >> 4;
  const unsigned short* p =
      base + (size_t)(row0 + (lane & 15)) * ld + k0 + half * 8;
  __builtin_prefetch(p + 32, 0, 1);           // next K tile
  union { v16bf v; uint4 q[2]; } f;
  f.q[0] = *reinterpret_cast<const uint4*>(p);
  f.q[1] = *reinterpret_cast<const uint4*>(p + 16);
  return f.v;
}

// Load one 16x32 bf16 fragment from an LDS tile (128 rows x 32 cols).
DEV_INLINE v16bf frag_from_lds(const unsigned short* tile, int row0) {
  const int lane = threadIdx.x & 31;
  const int half = lane >> 4;
  const unsigned short* p = tile + (row0 + (lane & 15)) * 32 + half * 8;
  union { v16bf v; uint4 q[2]; } f;
  f.q[0] = *reinterpret_cast<const uint4*>(p);
  f.q[1] = *reinterpret_cast<const uint4*>(p + 16);
  return f.v;
}

// Copy a 128x32 bf16 tile (8 KB) global -> LDS. 256 threads, 2 x 16B
// chunks each -> 2 global_load_async_to_lds_b128 per wave per tile.
DEV_INLINE void tile_copy_g2l(const unsigned short* g0, int ld, int row0, int k0,
                              unsigned short* l0) {
  const int tid = threadIdx.x;                // 0..255
#pragma unroll
  for (int i = 0; i < 2; ++i) {
    const int chunk = tid + i * 256;          // 0..511
    const int r  = chunk >> 2;
    const int c8 = (chunk & 3) * 8;           // bf16 offset of 16B chunk
    const unsigned short* g = g0 + (size_t)(row0 + r) * ld + k0 + c8;
    unsigned short* l = l0 + r * 32 + c8;
#if HAS_ASYNC
    __builtin_amdgcn_global_load_async_to_lds_b128(
        (__attribute__((address_space(1))) v4i*)g,
        (__attribute__((address_space(3))) v4i*)l, 0, 0);
#else
    *reinterpret_cast<uint4*>(l) = *reinterpret_cast<const uint4*>(g);
#endif
  }
}

// 128x128 block GEMM core with 8 waves (256 thr): wave (wave>>2, wave&3)
// computes a 64x32 tile -> acc[4][2]. smem: 4 x 4096 bf16 = 32 KB
// laid out as [buf][ A(128x32) | B(128x32) ].
DEV_INLINE void gemm_block_core(const unsigned short* __restrict__ A, int lda, int mblk,
                                const unsigned short* __restrict__ Bm, int ldb, int nblk,
                                int Ktot, unsigned short* smem, v8f acc[4][2]) {
  const int wave = threadIdx.x >> 5;
  const int wm   = (wave >> 2) * 64;          // wave row offset in block tile
  const int wn   = (wave & 3) * 32;           // wave col offset in block tile
  tile_copy_g2l(A,  lda, mblk, 0, smem);
  tile_copy_g2l(Bm, ldb, nblk, 0, smem + 4096);
  int buf = 0;
  for (int k0 = 0; k0 < Ktot; k0 += 32) {
    if (k0 + 32 < Ktot) {                     // prefetch next K tile
      unsigned short* nxt = smem + (buf ^ 1) * 8192;
      tile_copy_g2l(A,  lda, mblk, k0 + 32, nxt);
      tile_copy_g2l(Bm, ldb, nblk, k0 + 32, nxt + 4096);
      WAIT_ASYNC(4);                          // first 4 (current buf) done
    } else {
      WAIT_ASYNC(0);
    }
    __syncthreads();                          // publish current buffer
    const unsigned short* tA = smem + buf * 8192;
    const unsigned short* tB = tA + 4096;
    v16bf af[4];
#pragma unroll
    for (int ma = 0; ma < 4; ++ma) af[ma] = frag_from_lds(tA, wm + ma * 16);
#pragma unroll
    for (int nb = 0; nb < 2; ++nb) {
      v16bf bf = frag_from_lds(tB, wn + nb * 16);
#pragma unroll
      for (int ma = 0; ma < 4; ++ma)
        acc[ma][nb] = wmma_bf16(af[ma], bf, acc[ma][nb]);
    }
    __syncthreads();                          // done reading before rewrite
    buf ^= 1;
  }
}

// C/D fragment (16x16 f32): VGPR j holds row (j + 8*(lane>>4)), col = lane&15
DEV_INLINE void store_tile_f32(float* base, int ld, int row0, int col0, v8f a) {
  const int lane = threadIdx.x & 31;
  const int half = lane >> 4;
  const int col  = col0 + (lane & 15);
#pragma unroll
  for (int j = 0; j < 8; ++j)
    base[(size_t)(row0 + 8 * half + j) * ld + col] = a[j];
}

DEV_INLINE void store_tile_bf16(unsigned short* base, int ld, int row0, int col0, v8f a) {
  const int lane = threadIdx.x & 31;
  const int half = lane >> 4;
  const int col  = col0 + (lane & 15);
#pragma unroll
  for (int j = 0; j < 8; ++j)
    base[(size_t)(row0 + 8 * half + j) * ld + col] = f32_to_bf16(a[j]);
}

// ---------------- kernels -------------------------------------------

__global__ void cvt_weights(const float* __restrict__ Wq, const float* __restrict__ Wk,
                            const float* __restrict__ Wv, const float* __restrict__ Wo,
                            unsigned short* __restrict__ Wqkv,
                            unsigned short* __restrict__ WoB) {
  const int n1 = 3 * CI_ * C_;
  const int i  = blockIdx.x * blockDim.x + threadIdx.x;
  if (i < n1) {
    const int r = i / C_;
    float v = (r < CI_) ? Wq[i] : (r < 2 * CI_) ? Wk[i - CI_ * C_] : Wv[i - 2 * CI_ * C_];
    Wqkv[i] = f32_to_bf16(v);
  } else {
    WoB[i - n1] = f32_to_bf16(Wo[i - n1]);
  }
}

// x f32 [B][C][T] -> xT bf16 [B][T][C] via 32x32 LDS tile
__global__ void transpose_x(const float* __restrict__ x, unsigned short* __restrict__ xT) {
  __shared__ float tile[32][33];
  const int b  = blockIdx.z;
  const int c0 = blockIdx.y * 32;
  const int t0 = blockIdx.x * 32;
  const float* xb = x + (size_t)b * C_ * T_;
  for (int i = threadIdx.y; i < 32; i += 8)
    tile[i][threadIdx.x] = xb[(size_t)(c0 + i) * T_ + t0 + threadIdx.x];
  __syncthreads();
  unsigned short* ob = xT + (size_t)b * T_ * C_;
  for (int i = threadIdx.y; i < 32; i += 8)
    ob[(size_t)(t0 + i) * C_ + c0 + threadIdx.x] = f32_to_bf16(tile[threadIdx.x][i]);
}

// QKV projection: out[o,t] = sum_c Wqkv[o,c] * xT[b,t,c] + bias(o)
__global__ void __launch_bounds__(256, 1)
gemm_qkv(const unsigned short* __restrict__ W,
         const unsigned short* __restrict__ xT,
         const float* __restrict__ qb, const float* __restrict__ kb,
         const float* __restrict__ vb,
         unsigned short* __restrict__ Q, unsigned short* __restrict__ Kd,
         unsigned short* __restrict__ Vt) {
  __shared__ unsigned short smem[4 * 4096];
  const int b = blockIdx.z;
  v8f acc[4][2] = {};
  gemm_block_core(W, C_, blockIdx.y * 128,
                  xT + (size_t)b * T_ * C_, C_, blockIdx.x * 128, C_, smem, acc);
  const int wave = threadIdx.x >> 5;
  const int lane = threadIdx.x & 31;
  const int half = lane >> 4;
  const int m0 = blockIdx.y * 128 + (wave >> 2) * 64;
  const int n0 = blockIdx.x * 128 + (wave & 3) * 32;
  unsigned short* Qb = Q  + (size_t)b * T_ * CI_;
  unsigned short* Kb = Kd + (size_t)b * T_ * CI_;
  unsigned short* Vb = Vt + (size_t)b * CI_ * T_;
#pragma unroll
  for (int ma = 0; ma < 4; ++ma)
#pragma unroll
    for (int nb = 0; nb < 2; ++nb) {
      const int o_base = m0 + 16 * ma + 8 * half;
      const int t      = n0 + 16 * nb + (lane & 15);
#pragma unroll
      for (int j = 0; j < 8; ++j) {
        const int o = o_base + j;
        const float v = acc[ma][nb][j];
        if (o < CI_) {
          Qb[(size_t)t * CI_ + o] = f32_to_bf16(v + qb[o]);
        } else if (o < 2 * CI_) {
          Kb[(size_t)t * CI_ + (o - CI_)] = f32_to_bf16(v + kb[o - CI_]);
        } else {
          Vb[(size_t)(o - 2 * CI_) * T_ + t] = f32_to_bf16(v + vb[o - 2 * CI_]);
        }
      }
    }
}

// Scores: S[b][q][k] = sum_c Q[b][q][c] * K[b][k][c]   (f32 out)
__global__ void __launch_bounds__(256, 1)
gemm_scores(const unsigned short* __restrict__ Q,
            const unsigned short* __restrict__ K,
            float* __restrict__ S) {
  __shared__ unsigned short smem[4 * 4096];
  const int b = blockIdx.z;
  v8f acc[4][2] = {};
  gemm_block_core(Q + (size_t)b * T_ * CI_, CI_, blockIdx.y * 128,
                  K + (size_t)b * T_ * CI_, CI_, blockIdx.x * 128, CI_, smem, acc);
  const int wave = threadIdx.x >> 5;
  const int q0 = blockIdx.y * 128 + (wave >> 2) * 64;
  const int k0 = blockIdx.x * 128 + (wave & 3) * 32;
  float* Sb = S + (size_t)b * T_ * T_;
#pragma unroll
  for (int ma = 0; ma < 4; ++ma)
#pragma unroll
    for (int nb = 0; nb < 2; ++nb)
      store_tile_f32(Sb, T_, q0 + 16 * ma, k0 + 16 * nb, acc[ma][nb]);
}

// Column softmax stats: per (b,k) online max + sum of exp over q.
__global__ void col_stats(const float* __restrict__ S,
                          float* __restrict__ cmax, float* __restrict__ cinv) {
  const int idx = blockIdx.x * blockDim.x + threadIdx.x;  // over B*T
  const int b = idx / T_, k = idx % T_;
  const float* col = S + (size_t)b * T_ * T_ + k;
  float m = -3.0e38f, s = 0.f;
  for (int q = 0; q < T_; ++q) {
    const float v  = col[(size_t)q * T_];
    const float nm = fmaxf(m, v);
    s = s * __expf(m - nm) + __expf(v - nm);
    m = nm;
  }
  cmax[idx] = m;
  cinv[idx] = 1.0f / s;
}

// Build bf16 probability fragment from f32 scores + LDS column stats.
DEV_INLINE v16bf load_frag_P(const float* Sb, int row0, int k0,
                             const float* s_m, const float* s_r) {
  const int lane = threadIdx.x & 31;
  const int half = lane >> 4;
  const float* Srow = Sb + (size_t)(row0 + (lane & 15)) * T_ + k0 + half * 8;
  union { v16bf v; unsigned short h[16]; } f;
#pragma unroll
  for (int e = 0; e < 8; ++e) {
    const int k1 = k0 + half * 8 + e;
    const int k2 = k1 + 16;
    f.h[e]     = f32_to_bf16(__expf(Srow[e]      - s_m[k1]) * s_r[k1]);
    f.h[8 + e] = f32_to_bf16(__expf(Srow[16 + e] - s_m[k2]) * s_r[k2]);
  }
  return f.v;
}

// y[q,c] = sum_k P[q,k] * Vt[c,k]; stored Y [B][T][CI] bf16.
__global__ void __launch_bounds__(128, 1)
gemm_pv(const float* __restrict__ S,
        const float* __restrict__ cmax, const float* __restrict__ cinv,
        const unsigned short* __restrict__ Vt,
        unsigned short* __restrict__ Y) {
  __shared__ float s_m[T_];
  __shared__ float s_r[T_];
  const int b = blockIdx.z;
  for (int i = threadIdx.x; i < T_; i += blockDim.x) {
    s_m[i] = cmax[b * T_ + i];
    s_r[i] = cinv[b * T_ + i];
  }
  __syncthreads();
  const int wave = threadIdx.x >> 5;
  const int q0   = blockIdx.y * 64 + (wave >> 1) * 32;
  const int c0   = blockIdx.x * 64 + (wave & 1) * 32;
  const float* Sb          = S  + (size_t)b * T_ * T_;
  const unsigned short* Vb = Vt + (size_t)b * CI_ * T_;
  v8f acc00 = {}, acc01 = {}, acc10 = {}, acc11 = {};
  for (int k0 = 0; k0 < T_; k0 += 32) {
    v16bf a0 = load_frag_P(Sb, q0,      k0, s_m, s_r);
    v16bf a1 = load_frag_P(Sb, q0 + 16, k0, s_m, s_r);
    v16bf b0 = load_frag_b(Vb, T_, c0,      k0);
    v16bf b1 = load_frag_b(Vb, T_, c0 + 16, k0);
    acc00 = wmma_bf16(a0, b0, acc00);
    acc01 = wmma_bf16(a0, b1, acc01);
    acc10 = wmma_bf16(a1, b0, acc10);
    acc11 = wmma_bf16(a1, b1, acc11);
  }
  unsigned short* Yb = Y + (size_t)b * T_ * CI_;
  store_tile_bf16(Yb, CI_, q0,      c0,      acc00);
  store_tile_bf16(Yb, CI_, q0,      c0 + 16, acc01);
  store_tile_bf16(Yb, CI_, q0 + 16, c0,      acc10);
  store_tile_bf16(Yb, CI_, q0 + 16, c0 + 16, acc11);
}

// wy[o,t] = sum_c Wo[o,c] * Y[b,t,c] + Wo_b[o]   (f32 out, [B][C][T])
__global__ void __launch_bounds__(256, 1)
gemm_out(const unsigned short* __restrict__ Wo,
         const unsigned short* __restrict__ Y,
         const float* __restrict__ ob, float* __restrict__ WY) {
  __shared__ unsigned short smem[4 * 4096];
  const int b = blockIdx.z;
  v8f acc[4][2] = {};
  gemm_block_core(Wo, CI_, blockIdx.y * 128,
                  Y + (size_t)b * T_ * CI_, CI_, blockIdx.x * 128, CI_, smem, acc);
  const int wave = threadIdx.x >> 5;
  const int lane = threadIdx.x & 31;
  const int half = lane >> 4;
  const int m0 = blockIdx.y * 128 + (wave >> 2) * 64;
  const int n0 = blockIdx.x * 128 + (wave & 3) * 32;
  float* Wb = WY + (size_t)b * C_ * T_;
#pragma unroll
  for (int ma = 0; ma < 4; ++ma)
#pragma unroll
    for (int nb = 0; nb < 2; ++nb) {
      const int o_base = m0 + 16 * ma + 8 * half;
      const int t      = n0 + 16 * nb + (lane & 15);
#pragma unroll
      for (int j = 0; j < 8; ++j)
        Wb[(size_t)(o_base + j) * T_ + t] = acc[ma][nb][j] + ob[o_base + j];
    }
}

// Per-channel mean / rstd over (B,T)
__global__ void bn_stats(const float* __restrict__ WY,
                         float* __restrict__ mean, float* __restrict__ rstd) {
  const int c   = blockIdx.x;
  const int tid = threadIdx.x;
  float s = 0.f, ss = 0.f;
  for (int i = tid; i < B_ * T_; i += blockDim.x) {
    const int b = i >> 11;
    const int t = i & (T_ - 1);
    const float v = WY[((size_t)b * C_ + c) * T_ + t];
    s += v; ss += v * v;
  }
  __shared__ float rs[256], rq[256];
  rs[tid] = s; rq[tid] = ss;
  __syncthreads();
  for (int off = 128; off > 0; off >>= 1) {
    if (tid < off) { rs[tid] += rs[tid + off]; rq[tid] += rq[tid + off]; }
    __syncthreads();
  }
  if (tid == 0) {
    const float n = (float)(B_ * T_);
    const float m = rs[0] / n;
    const float var = rq[0] / n - m * m;
    mean[c] = m;
    rstd[c] = rsqrtf(var + BN_EPS_);
  }
}

// out = (wy - mean)*rstd*gamma + beta + x
__global__ void bn_apply(const float* __restrict__ WY, const float* __restrict__ x,
                         const float* __restrict__ mean, const float* __restrict__ rstd,
                         const float* __restrict__ gamma, const float* __restrict__ beta,
                         float* __restrict__ out) {
  const size_t i = (size_t)blockIdx.x * blockDim.x + threadIdx.x;
  const int c = (int)((i / T_) % C_);
  out[i] = (WY[i] - mean[c]) * rstd[c] * gamma[c] + beta[c] + x[i];
}

// ---------------- launch --------------------------------------------
extern "C" void kernel_launch(void* const* d_in, const int* in_sizes, int n_in,
                              void* d_out, int out_size, void* d_ws, size_t ws_size,
                              hipStream_t stream) {
  (void)in_sizes; (void)n_in; (void)out_size; (void)ws_size;
  const float* x    = (const float*)d_in[0];
  const float* Wq_w = (const float*)d_in[1];
  const float* Wq_b = (const float*)d_in[2];
  const float* Wk_w = (const float*)d_in[3];
  const float* Wk_b = (const float*)d_in[4];
  const float* Wv_w = (const float*)d_in[5];
  const float* Wv_b = (const float*)d_in[6];
  const float* Wo_w = (const float*)d_in[7];
  const float* Wo_b = (const float*)d_in[8];
  const float* gamma = (const float*)d_in[9];
  const float* beta  = (const float*)d_in[10];
  float* out = (float*)d_out;

  char* ws = (char*)d_ws;
  unsigned short* xT   = (unsigned short*)(ws + OFF_XT);
  unsigned short* Wqkv = (unsigned short*)(ws + OFF_WQKV);
  unsigned short* WoB  = (unsigned short*)(ws + OFF_WO);
  unsigned short* Q    = (unsigned short*)(ws + OFF_Q);
  unsigned short* K    = (unsigned short*)(ws + OFF_K);
  unsigned short* Vt   = (unsigned short*)(ws + OFF_VT);
  float*          S    = (float*)(ws + OFF_S);
  float*          cmax = (float*)(ws + OFF_CMAX);
  float*          cinv = (float*)(ws + OFF_CINV);
  unsigned short* Y    = (unsigned short*)(ws + OFF_Y);
  float*          WY   = (float*)(ws + OFF_WY);
  float*          mean = (float*)(ws + OFF_MEAN);
  float*          rstd = (float*)(ws + OFF_RSTD);

  cvt_weights<<<(3 * CI_ * C_ + C_ * CI_) / 256, 256, 0, stream>>>(
      Wq_w, Wk_w, Wv_w, Wo_w, Wqkv, WoB);
  transpose_x<<<dim3(T_ / 32, C_ / 32, B_), dim3(32, 8), 0, stream>>>(x, xT);
  // QKV projection (M=1536, N=2048, K=1024 per batch), 128x128 blocks
  gemm_qkv<<<dim3(T_ / 128, (3 * CI_) / 128, B_), 256, 0, stream>>>(
      Wqkv, xT, Wq_b, Wk_b, Wv_b, Q, K, Vt);
  // scores S = Q K^T (M=N=2048, K=512 per batch)
  gemm_scores<<<dim3(T_ / 128, T_ / 128, B_), 256, 0, stream>>>(Q, K, S);
  // column (query-axis) softmax stats
  col_stats<<<(B_ * T_) / 256, 256, 0, stream>>>(S, cmax, cinv);
  // Y = softmax_q(S) * V (M=2048, N=512, K=2048 per batch)
  gemm_pv<<<dim3(CI_ / 64, T_ / 64, B_), 128, 0, stream>>>(S, cmax, cinv, Vt, Y);
  // output projection (M=1024, N=2048, K=512 per batch)
  gemm_out<<<dim3(T_ / 128, C_ / 128, B_), 256, 0, stream>>>(WoB, Y, Wo_b, WY);
  // BatchNorm (training stats) + affine + residual
  bn_stats<<<C_, 256, 0, stream>>>(WY, mean, rstd);
  bn_apply<<<(size_t)B_ * C_ * T_ / 256, 256, 0, stream>>>(
      WY, x, mean, rstd, gamma, beta, out);
}
